// ConditionedAttention_70798240907872
// MI455X (gfx1250) — matrix operand order repaired and checked by reading, not compiled, not run on hardware
//
#include <hip/hip_runtime.h>

// Fused conditioned attention for MI455X (gfx1250, wave32, WMMA).
// One workgroup per (b,t,h); 8 waves x 16 rows = full 128-row tile.
// GEMM1: scores = Q(128x64) @ K1^T  via v_wmma_f32_16x16x32_f16
// GEMM2: out    = softmax(scores*s + log(OT+eps)) @ V1(128x64)

typedef __attribute__((ext_vector_type(16))) _Float16 v16h;
typedef __attribute__((ext_vector_type(8)))  _Float16 v8h;
typedef __attribute__((ext_vector_type(4)))  _Float16 v4h;
typedef __attribute__((ext_vector_type(8)))  float    v8f;
typedef __attribute__((ext_vector_type(4)))  float    v4f;

namespace {
constexpr int T_DIM = 24;
constexpr int S_DIM = 128;
constexpr int H_DIM = 8;
constexpr int D_DIM = 64;
constexpr int HD    = H_DIM * D_DIM;     // row stride (elements) in Q/K/V
constexpr int NWAVE = 8;
constexpr int KSTR  = 72;    // K-tile LDS row stride in halves (144B, 16B aligned, conflict-free)
constexpr int PSTR  = 136;   // P row stride in halves (272B)
constexpr int VTSTR = 136;   // Vt row stride in halves (272B)
constexpr int REGION0 = NWAVE * 16 * PSTR;   // 17408 halves; aliases K tile (128*72 = 9216 halves)
constexpr int REGION1 = D_DIM * VTSTR;       // 8704 halves
// total LDS = (17408 + 8704) * 2 = 52224 bytes
}

__global__ __launch_bounds__(256)
void ca_fused_attention(const float* __restrict__ Q, const float* __restrict__ K,
                        const float* __restrict__ V, const float* __restrict__ OT,
                        float* __restrict__ Out) {
  __shared__ _Float16 lds[REGION0 + REGION1];
  _Float16* ldsK  = lds;              // phase 1: K tile (f16, row-major, padded)
  _Float16* ldsP  = lds;              // phase 2: per-wave P staging (aliases K)
  _Float16* ldsVt = lds + REGION0;    // V tile transposed: Vt[d][z]

  const int blk = blockIdx.x;
  const int h   = blk % H_DIM;
  const int t   = (blk / H_DIM) % T_DIM;
  const int b   = blk / (H_DIM * T_DIM);
  const int t1  = (t + 1) % T_DIM;    // roll(K/V, -1, axis=T)

  const int tid  = threadIdx.x;
  const int lane = tid & 31;
  const int wave = tid >> 5;
  const int hs   = lane >> 4;         // half-wave select (0/1)
  const int lr   = lane & 15;

  const size_t base_q  = (((size_t)b * T_DIM + t ) * S_DIM) * HD + (size_t)h * D_DIM;
  const size_t base_kv = (((size_t)b * T_DIM + t1) * S_DIM) * HD + (size_t)h * D_DIM;
  const size_t base_ot = (((size_t)b * T_DIM + t ) * S_DIM) * (size_t)S_DIM;

  // ---- cooperative load: K tile (128x64 f32) -> f16 LDS, padded rows ----
  #pragma unroll
  for (int it = 0; it < 8; ++it) {
    int linear = (tid + it * 256) * 4;          // element index in 128x64 tile
    int row = linear >> 6;
    int col = linear & 63;
    v4f f = *(const v4f*)(K + base_kv + (size_t)row * HD + col);
    v4h hh;
    #pragma unroll
    for (int i = 0; i < 4; ++i) hh[i] = (_Float16)f[i];
    *(v4h*)(ldsK + row * KSTR + col) = hh;
  }
  // ---- cooperative load: V tile -> f16 LDS, transposed Vt[d][z] ----
  #pragma unroll
  for (int it = 0; it < 8; ++it) {
    int linear = (tid + it * 256) * 4;
    int z  = linear >> 6;
    int d0 = linear & 63;
    v4f f = *(const v4f*)(V + base_kv + (size_t)z * HD + d0);
    #pragma unroll
    for (int i = 0; i < 4; ++i) ldsVt[(d0 + i) * VTSTR + z] = (_Float16)f[i];
  }

  // ---- Q A-fragments (16x32 f16 layout): lane = M row, K chunks per half-wave ----
  const int m0 = wave * 16;
  const float* qrow = Q + base_q + (size_t)(m0 + lr) * HD;
  v16h afrag[2];
  #pragma unroll
  for (int kk = 0; kk < 2; ++kk) {
    int c0 = kk * 32 + 8 * hs;                  // K chunk start per ISA A layout
    #pragma unroll
    for (int hc = 0; hc < 2; ++hc) {
      int c = c0 + 16 * hc;
      v4f f0 = *(const v4f*)(qrow + c);
      v4f f1 = *(const v4f*)(qrow + c + 4);
      #pragma unroll
      for (int i = 0; i < 4; ++i) {
        afrag[kk][hc * 8 + i]     = (_Float16)f0[i];
        afrag[kk][hc * 8 + 4 + i] = (_Float16)f1[i];
      }
    }
  }

  __syncthreads();

  // ---- GEMM1: scores(16x128) = Q(16x64) @ K1^T, 8 z-tiles x 2 k-steps ----
  v8f acc[8];
  #pragma unroll
  for (int j = 0; j < 8; ++j) {
    #pragma unroll
    for (int r = 0; r < 8; ++r) acc[j][r] = 0.f;
  }
  #pragma unroll
  for (int kk = 0; kk < 2; ++kk) {
    #pragma unroll
    for (int j = 0; j < 8; ++j) {
      const _Float16* kb = ldsK + (j * 16 + lr) * KSTR + kk * 32 + 8 * hs;
      union { v16h v; v8h h[2]; } bf;
      bf.h[0] = *(const v8h*)(kb);
      bf.h[1] = *(const v8h*)(kb + 16);
      acc[j] = __builtin_amdgcn_wmma_f32_16x16x32_f16(false, afrag[kk], false, bf.v,
                                                      (short)0, acc[j], false, false);
    }
  }

  // ---- scale + log(OT + eps); C layout: lane holds row (r + 8*hs), col (16j + lr) ----
  #pragma unroll
  for (int j = 0; j < 8; ++j) {
    #pragma unroll
    for (int r = 0; r < 8; ++r) {
      int srow = m0 + r + 8 * hs;
      float o = OT[base_ot + (size_t)srow * S_DIM + j * 16 + lr];
      acc[j][r] = acc[j][r] * 0.125f + __logf(o + 1e-8f);
    }
  }

  // ---- row softmax: each row lives in one 16-lane half across 8 tile VGPRs ----
  float rinv[8];
  #pragma unroll
  for (int r = 0; r < 8; ++r) {
    float m = acc[0][r];
    #pragma unroll
    for (int j = 1; j < 8; ++j) m = fmaxf(m, acc[j][r]);
    #pragma unroll
    for (int off = 1; off < 16; off <<= 1) m = fmaxf(m, __shfl_xor(m, off, 32));
    #pragma unroll
    for (int j = 0; j < 8; ++j) acc[j][r] = __expf(acc[j][r] - m);
    float s = 0.f;
    #pragma unroll
    for (int j = 0; j < 8; ++j) s += acc[j][r];
    #pragma unroll
    for (int off = 1; off < 16; off <<= 1) s += __shfl_xor(s, off, 32);
    rinv[r] = __builtin_amdgcn_rcpf(s);   // fold 1/rowsum into output store
  }

  __syncthreads();   // all waves done reading ldsK -> safe to alias it with P

  // ---- stage P (f16) in per-wave LDS region (C layout -> row-major) ----
  _Float16* myP = ldsP + wave * (16 * PSTR);
  #pragma unroll
  for (int j = 0; j < 8; ++j) {
    #pragma unroll
    for (int r = 0; r < 8; ++r)
      myP[(r + 8 * hs) * PSTR + j * 16 + lr] = (_Float16)acc[j][r];
  }

  __syncthreads();

  // ---- GEMM2: out(16x64) = P(16x128) @ V(128x64), 4 k-steps x 4 d-tiles ----
  v8f oacc[4];
  #pragma unroll
  for (int n = 0; n < 4; ++n) {
    #pragma unroll
    for (int r = 0; r < 8; ++r) oacc[n][r] = 0.f;
  }
  #pragma unroll
  for (int kk = 0; kk < 4; ++kk) {
    const _Float16* pr = myP + lr * PSTR + kk * 32 + 8 * hs;   // A layout chunks
    union { v16h v; v8h h[2]; } pa;
    pa.h[0] = *(const v8h*)(pr);
    pa.h[1] = *(const v8h*)(pr + 16);
    #pragma unroll
    for (int n = 0; n < 4; ++n) {
      const _Float16* vb = ldsVt + (n * 16 + lr) * VTSTR + kk * 32 + 8 * hs;
      union { v16h v; v8h h[2]; } bf;
      bf.h[0] = *(const v8h*)(vb);
      bf.h[1] = *(const v8h*)(vb + 16);
      oacc[n] = __builtin_amdgcn_wmma_f32_16x16x32_f16(false, pa.v, false, bf.v,
                                                       (short)0, oacc[n], false, false);
    }
  }

  // ---- normalize by 1/rowsum and store (coalesced 64B per half-wave) ----
  #pragma unroll
  for (int r = 0; r < 8; ++r) {
    int srow = m0 + r + 8 * hs;
    float* orow = Out + base_q + (size_t)srow * HD;
    #pragma unroll
    for (int n = 0; n < 4; ++n) orow[n * 16 + lr] = oacc[n][r] * rinv[r];
  }
}

extern "C" void kernel_launch(void* const* d_in, const int* in_sizes, int n_in,
                              void* d_out, int out_size, void* d_ws, size_t ws_size,
                              hipStream_t stream) {
  const float* Q  = (const float*)d_in[0];
  const float* K  = (const float*)d_in[1];
  const float* V  = (const float*)d_in[2];
  const float* OT = (const float*)d_in[3];
  float* Out = (float*)d_out;
  (void)n_in; (void)out_size; (void)d_ws; (void)ws_size;

  const int B = in_sizes[0] / (T_DIM * S_DIM * H_DIM * D_DIM);
  dim3 grid(B * T_DIM * H_DIM);   // one workgroup per (b,t,h) = 1536 for B=8
  ca_fused_attention<<<grid, dim3(256), 0, stream>>>(Q, K, V, OT, Out);
}
